// GraphConvLayerMat_32495722561789
// MI455X (gfx1250) — matrix operand order repaired and checked by reading, not compiled, not run on hardware
//
#include <hip/hip_runtime.h>
#include <hip/hip_fp16.h>
#include <math.h>

typedef __attribute__((ext_vector_type(16))) _Float16 v16h;
typedef __attribute__((ext_vector_type(8)))  float    v8f;

#define DFEAT 128
#define HID   128
#define BN_EPS 1e-3f

// ---------------------------------------------------------------------------
// Kernel 1: zero the accumulation buffer h  (float4 grid-stride memset)
// ---------------------------------------------------------------------------
__global__ void k_zero_f4(float4* __restrict__ p, long long n4) {
    long long i = (long long)blockIdx.x * blockDim.x + threadIdx.x;
    long long stride = (long long)gridDim.x * blockDim.x;
    float4 z = make_float4(0.f, 0.f, 0.f, 0.f);
    for (; i < n4; i += stride) p[i] = z;
}

// ---------------------------------------------------------------------------
// Kernel 2: SpMM scatter  h[dst] += w * H[src]
// One wave32 per edge: lane l handles features [4l, 4l+4) -> one float4 load
// from H[src] (512B row, fully coalesced across the wave) + 4 f32 atomics
// that stay resident in L2 (h = 51.2MB < 192MB).
// ---------------------------------------------------------------------------
__global__ void k_spmm(const float* __restrict__ H,
                       const int*   __restrict__ src,
                       const int*   __restrict__ dst,
                       const float* __restrict__ w,
                       float*       __restrict__ h,
                       int E) {
    int lane   = threadIdx.x & 31;
    int wave   = (int)((blockIdx.x * blockDim.x + threadIdx.x) >> 5);
    int nwaves = (int)((gridDim.x * blockDim.x) >> 5);
    for (int e = wave; e < E; e += nwaves) {
        int   s  = src[e];
        int   d  = dst[e];
        float we = w[e];
        float4 v = *(const float4*)(H + (size_t)s * DFEAT + lane * 4);
        float* o = h + (size_t)d * DFEAT + lane * 4;
        atomicAdd(o + 0, v.x * we);
        atomicAdd(o + 1, v.y * we);
        atomicAdd(o + 2, v.z * we);
        atomicAdd(o + 3, v.w * we);
    }
}

// ---------------------------------------------------------------------------
// Kernel 3: fold BatchNorm into the dense layer (constant folding).
//   s[k] = gamma[k]*rsqrt(var[k]+eps);  t[k] = beta[k] - mean[k]*s[k]
//   W'[k][n] = s[k]*W[k][n]   (stored transposed, f16, for WMMA B layout)
//   b'[n]    = b[n] + sum_k t[k]*W[k][n]
// One block of 128 threads; thread n owns output column n.
// ---------------------------------------------------------------------------
__global__ void k_fold_bn(const float* __restrict__ W,
                          const float* __restrict__ b,
                          const float* __restrict__ gamma,
                          const float* __restrict__ beta,
                          const float* __restrict__ mean,
                          const float* __restrict__ var,
                          _Float16*    __restrict__ WT,   // [HID][DFEAT] f16
                          float*       __restrict__ bp) { // [HID]
    __shared__ float s_s[DFEAT];
    __shared__ float s_t[DFEAT];
    int t = threadIdx.x;                       // 0..127
    float sc = gamma[t] * rsqrtf(var[t] + BN_EPS);
    s_s[t] = sc;
    s_t[t] = beta[t] - mean[t] * sc;
    __syncthreads();
    int n = t;
    float acc = b[n];
    for (int k = 0; k < DFEAT; ++k) {
        float wkn = W[k * HID + n];
        acc += s_t[k] * wkn;
        WT[(size_t)n * DFEAT + k] = (_Float16)(s_s[k] * wkn);
    }
    bp[n] = acc;
}

// ---------------------------------------------------------------------------
// Kernel 4: out = GELU_exact( h @ W' + b' )  via v_wmma_f32_16x16x32_f16.
// One wave32 per 16-row strip. A panel (16x128) loaded f32 and converted to
// f16 in registers (lanes 0-15: K {0..15}; lanes
// 16-31: K {16..31}); held across all 8 N-tiles. B = W'^T, each lane
// reads 32 contiguous bytes (K-run for its column). C layout: M=(lane>>4)*8+i.
// ---------------------------------------------------------------------------
__global__ void k_gemm_gelu(const float*    __restrict__ h,
                            const _Float16* __restrict__ WT,
                            const float*    __restrict__ bp,
                            float*          __restrict__ out,
                            int N) {
    int lane = threadIdx.x & 31;
    int wave = (int)((blockIdx.x * blockDim.x + threadIdx.x) >> 5);
    int ntiles = (N + 15) >> 4;
    if (wave >= ntiles) return;

    int r0 = wave * 16;
    int lr = lane & 15;
    int hi = lane >> 4;

    int row = r0 + lr;
    if (row >= N) row = N - 1;           // clamp for (unused here) tail tiles
    const float* arow = h + (size_t)row * DFEAT;

    v16h A[4];
#pragma unroll
    for (int t = 0; t < 4; ++t) {
        int kb = t * 32 + hi * 16;
        float4 p0 = *(const float4*)(arow + kb);
        float4 p1 = *(const float4*)(arow + kb + 4);
        float4 p2 = *(const float4*)(arow + kb + 8);
        float4 p3 = *(const float4*)(arow + kb + 12);
        A[t][0]  = (_Float16)p0.x; A[t][1]  = (_Float16)p0.y;
        A[t][2]  = (_Float16)p0.z; A[t][3]  = (_Float16)p0.w;
        A[t][4]  = (_Float16)p1.x; A[t][5]  = (_Float16)p1.y;
        A[t][6]  = (_Float16)p1.z; A[t][7]  = (_Float16)p1.w;
        A[t][8]  = (_Float16)p2.x; A[t][9]  = (_Float16)p2.y;
        A[t][10] = (_Float16)p2.z; A[t][11] = (_Float16)p2.w;
        A[t][12] = (_Float16)p3.x; A[t][13] = (_Float16)p3.y;
        A[t][14] = (_Float16)p3.z; A[t][15] = (_Float16)p3.w;
    }

#pragma unroll
    for (int nt = 0; nt < 8; ++nt) {
        int n = nt * 16 + lr;
        float bb = bp[n];
        v8f c;
#pragma unroll
        for (int i = 0; i < 8; ++i) c[i] = bb;   // bias depends on N only

        const _Float16* bcol = WT + (size_t)n * DFEAT + hi * 16;
#pragma unroll
        for (int t = 0; t < 4; ++t) {
            v16h B = *(const v16h*)(bcol + t * 32);
            c = __builtin_amdgcn_wmma_f32_16x16x32_f16(
                    false, A[t], false, B, (short)0, c, false, false);
        }

#pragma unroll
        for (int i = 0; i < 8; ++i) {
            int m = r0 + hi * 8 + i;
            if (m < N) {
                float x = c[i];
                float g = 0.5f * x * (1.0f + erff(x * 0.70710678118654752f));
                out[(size_t)m * HID + n] = g;
            }
        }
    }
}

// ---------------------------------------------------------------------------
// Launch
// inputs: 0:H 1:edge_src 2:edge_dst 3:edge_w 4:gamma 5:beta 6:moving_mean
//         7:moving_var 8:W 9:b 10:num_nodes
// ---------------------------------------------------------------------------
extern "C" void kernel_launch(void* const* d_in, const int* in_sizes, int n_in,
                              void* d_out, int out_size, void* d_ws, size_t ws_size,
                              hipStream_t stream) {
    const float* H     = (const float*)d_in[0];
    const int*   esrc  = (const int*)  d_in[1];
    const int*   edst  = (const int*)  d_in[2];
    const float* ew    = (const float*)d_in[3];
    const float* gamma = (const float*)d_in[4];
    const float* beta  = (const float*)d_in[5];
    const float* mean  = (const float*)d_in[6];
    const float* var   = (const float*)d_in[7];
    const float* W     = (const float*)d_in[8];
    const float* b     = (const float*)d_in[9];

    int N = in_sizes[0] / DFEAT;   // 100000
    int E = in_sizes[1];           // 3200000

    // workspace layout (d_ws): h [N*128 f32] | WT' [128*128 f16] | b' [128 f32]
    char* ws = (char*)d_ws;
    float*    h_acc = (float*)ws;
    size_t    h_bytes = (size_t)N * DFEAT * sizeof(float);
    h_bytes = (h_bytes + 255) & ~(size_t)255;
    _Float16* WT = (_Float16*)(ws + h_bytes);
    float*    bp = (float*)(ws + h_bytes + (size_t)HID * DFEAT * sizeof(_Float16));
    float*    out = (float*)d_out;

    // 1) zero h
    long long n4 = (long long)N * DFEAT / 4;
    k_zero_f4<<<2048, 256, 0, stream>>>((float4*)h_acc, n4);

    // 2) fold BN into weights/bias (tiny, independent of h)
    k_fold_bn<<<1, 128, 0, stream>>>(W, b, gamma, beta, mean, var, WT, bp);

    // 3) SpMM scatter: 32768 waves, ~98 edges each
    k_spmm<<<4096, 256, 0, stream>>>(H, esrc, edst, ew, h_acc, E);

    // 4) WMMA GEMM + exact GELU: one wave per 16-row tile
    int ntiles = (N + 15) / 16;
    int blocks = (ntiles * 32 + 255) / 256;
    k_gemm_gelu<<<blocks, 256, 0, stream>>>(h_acc, WT, bp, out, N);
}
